// GCN_model1_23081154249329
// MI455X (gfx1250) — matrix operand order repaired and checked by reading, not compiled, not run on hardware
//
#include <hip/hip_runtime.h>

// ---------- types ----------
typedef __attribute__((ext_vector_type(16))) __bf16 v16bf;
typedef __attribute__((ext_vector_type(8)))  float  v8f;

struct U8 { unsigned int u[8]; };

__device__ __forceinline__ unsigned short f2bfbits(float f) {
  unsigned int u = __float_as_uint(f);
  u += 0x7FFFu + ((u >> 16) & 1u);          // round-to-nearest-even
  return (unsigned short)(u >> 16);
}
__device__ __forceinline__ __bf16 f2bf(float f) {
  unsigned short h = f2bfbits(f);
  return __builtin_bit_cast(__bf16, h);
}

// ---------------- B-fragment packing ----------------
// Pack W [K x 128] (f32, row-major K x N) into wave32 WMMA B fragments.
// B 32x16 bf16 layout (contiguous k order used here, same as A; not ISA table order): lanes 0-15 hold K=0..15
// (VGPR j -> K=2j,2j+1 packed in one dword), lanes 16-31 hold K=16..31.
// One thread per (ktile, ntile, lane); fragment stored lane-contiguous so the
// GEMM reads it with two coalesced uint4 loads.
__global__ void gcn_pack_B(const float* __restrict__ W,
                           unsigned int* __restrict__ Bfrag, int K) {
  int idx = blockIdx.x * blockDim.x + threadIdx.x;
  int total = (K / 32) * 8 * 32;
  if (idx >= total) return;
  int lane = idx & 31;
  int nt   = (idx >> 5) & 7;
  int kt   = idx >> 8;
  int half = lane >> 4;
  int col  = nt * 16 + (lane & 15);
#pragma unroll
  for (int j = 0; j < 8; ++j) {
    int kk = kt * 32 + half * 16 + 2 * j;
    unsigned int lo = f2bfbits(W[(size_t)kk * 128 + col]);
    unsigned int hi = f2bfbits(W[(size_t)(kk + 1) * 128 + col]);
    Bfrag[(size_t)idx * 8 + j] = lo | (hi << 16);
  }
}

// ---------------- WMMA GEMM: C[M x 128] = A[M x K] * W[K x 128] ----------------
// 256 threads = 8 waves; block computes 128 rows; each wave computes a
// 16(M) x 128(N) strip = 8 f32 accumulators. A tile staged via LDS in bf16.
// f32 A path: coalesced float4 loads + in-register bf16 convert -> LDS.
// bf16 A path: CDNA5 GLOBAL_LOAD_ASYNC_TO_LDS_B128 (ASYNCcnt-tracked DMA).
template <int K, bool A_IS_F32>
__global__ __launch_bounds__(256) void gcn_gemm_wmma(
    const void* __restrict__ Aptr, const unsigned int* __restrict__ Bfrag,
    float* __restrict__ C, int M) {
  constexpr int TILES = K / 32;
  // stride 40 elems = 80B rows: 16B-aligned (async B128 dest) and
  // bank-conflict-free for 16-row fragment reads (dword stride 20).
  __shared__ __bf16 Alds[128][40];
  const float*  Af = (const float*)Aptr;
  const __bf16* Ab = (const __bf16*)Aptr;

  int tid  = threadIdx.x;
  int w    = tid >> 5, lane = tid & 31;
  int half = lane >> 4, r = lane & 15;
  int baseRow = blockIdx.x * 128;

  v8f acc[8];
#pragma unroll
  for (int i = 0; i < 8; ++i) {
    v8f z = {0.f, 0.f, 0.f, 0.f, 0.f, 0.f, 0.f, 0.f};
    acc[i] = z;
  }

  // cooperative staging mapping: 2 threads per row, 16 elems (32B bf16) each
  int srow = tid >> 1;
  int scol = (tid & 1) * 16;
  int gr = baseRow + srow; if (gr > M - 1) gr = M - 1;

  for (int kt = 0; kt < TILES; ++kt) {
    __syncthreads();
    if constexpr (A_IS_F32) {
      const float* ap = Af + (size_t)gr * K + kt * 32 + scol;
      if (kt + 1 < TILES) __builtin_prefetch(ap + 32, 0, 1);
#pragma unroll
      for (int q = 0; q < 4; ++q) {
        float4 v = *(const float4*)(ap + 4 * q);
        Alds[srow][scol + 4 * q + 0] = f2bf(v.x);
        Alds[srow][scol + 4 * q + 1] = f2bf(v.y);
        Alds[srow][scol + 4 * q + 2] = f2bf(v.z);
        Alds[srow][scol + 4 * q + 3] = f2bf(v.w);
      }
    } else {
      // Async DMA: 32 contiguous bytes per thread, offset applies to both
      // the global and LDS side (ISA 10.7 pseudocode). Tracked by ASYNCcnt.
      const char* gp = (const char*)(Ab + (size_t)gr * K + kt * 32 + scol);
      unsigned lds0 = (unsigned)(unsigned long long)&Alds[srow][scol];
      asm volatile("global_load_async_to_lds_b128 %0, %1, off"
                   :: "v"(lds0), "v"(gp) : "memory");
      asm volatile("global_load_async_to_lds_b128 %0, %1, off offset:16"
                   :: "v"(lds0), "v"(gp) : "memory");
      asm volatile("s_wait_asynccnt 0x0" ::: "memory");
    }
    __syncthreads();

    // A fragment, 16x32 bf16 layout (matches B packing):
    // lanes 0-15: V0-7 -> K=0..15 ; lanes 16-31: K=16..31
    v16bf af;
#pragma unroll
    for (int j = 0; j < 8; ++j) {
      int kk = half * 16 + 2 * j;
      af[2 * j]     = Alds[w * 16 + r][kk];
      af[2 * j + 1] = Alds[w * 16 + r][kk + 1];
    }

    // Batch-load all 8 B fragments (one long load clause), then run the
    // WMMA chain back-to-back instead of load->wait->wmma per tile.
    U8 bt[8];
#pragma unroll
    for (int nt = 0; nt < 8; ++nt) {
      const unsigned int* bp =
          Bfrag + (((size_t)kt * 8 + nt) * 32 + lane) * 8;
#pragma unroll
      for (int q = 0; q < 8; ++q) bt[nt].u[q] = bp[q];
    }
#pragma unroll
    for (int nt = 0; nt < 8; ++nt) {
      v16bf bf_ = __builtin_bit_cast(v16bf, bt[nt]);
      acc[nt] = __builtin_amdgcn_wmma_f32_16x16x32_bf16(
          false, af, false, bf_, (short)0, acc[nt], false, false);
    }
  }

  // C/D layout: lanes 0-15 -> M = v, N = lane ; lanes 16-31 -> M = 8+v
#pragma unroll
  for (int nt = 0; nt < 8; ++nt) {
#pragma unroll
    for (int v = 0; v < 8; ++v) {
      int grow = baseRow + w * 16 + half * 8 + v;
      if (grow < M) C[(size_t)grow * 128 + nt * 16 + r] = acc[nt][v];
    }
  }
}

// ---------------- degree / dinv ----------------
__global__ void gcn_deg_init(float* deg, int n) {
  int i = blockIdx.x * blockDim.x + threadIdx.x;
  if (i < n) deg[i] = 1.0f;   // self loop
}
__global__ void gcn_deg_count(const int* __restrict__ dst, float* deg, int E) {
  int e = blockIdx.x * blockDim.x + threadIdx.x;
  if (e < E) atomicAdd(&deg[dst[e]], 1.0f);
}
__global__ void gcn_dinv(const float* __restrict__ deg, float* dinv, int n) {
  int i = blockIdx.x * blockDim.x + threadIdx.x;
  if (i < n) dinv[i] = rsqrtf(deg[i]);   // deg >= 1 always (self loops)
}

// ---------------- edge scatter (one wave per edge, 128 feats) ----------------
__global__ __launch_bounds__(256) void gcn_edge_scatter(
    const float* __restrict__ h, const float* __restrict__ dinv,
    const int* __restrict__ src, const int* __restrict__ dst,
    float* __restrict__ out, int E) {
  int e = blockIdx.x * 8 + (threadIdx.x >> 5);
  int lane = threadIdx.x & 31;
  if (e >= E) return;
  int s = src[e], d = dst[e];
  float norm = dinv[s] * dinv[d];
  float4 v = ((const float4*)(h + (size_t)s * 128))[lane];  // L2-resident gather
  float* o = out + (size_t)d * 128 + lane * 4;
  atomicAdd(o + 0, v.x * norm);
  atomicAdd(o + 1, v.y * norm);
  atomicAdd(o + 2, v.z * norm);
  atomicAdd(o + 3, v.w * norm);
}

// ---------------- self-loop + bias + relu (optionally recode bf16) ----------------
__global__ void gcn_post(const float* __restrict__ agg, const float* __restrict__ h,
                         const float* __restrict__ dinv, const float* __restrict__ bias,
                         __bf16* obf, float* of32) {
  int n = blockIdx.x, f = threadIdx.x;
  float di = dinv[n];
  size_t i = (size_t)n * 128 + f;
  float v = agg[i] + h[i] * di * di + bias[f];
  v = fmaxf(v, 0.0f);
  if (obf)  obf[i]  = f2bf(v);
  if (of32) of32[i] = v;
}

// ---------------- mean pool (sum + count) ----------------
__global__ void gcn_pool(const float* __restrict__ h, const int* __restrict__ batch,
                         float* __restrict__ gsum, float* __restrict__ cnt) {
  int n = blockIdx.x, f = threadIdx.x;
  int g = batch[n];
  atomicAdd(&gsum[(size_t)g * 128 + f], h[(size_t)n * 128 + f]);
  if (f == 0) atomicAdd(&cnt[g], 1.0f);
}

// ---------------- classifier: out[64x20] = (gsum/cnt) @ Wc + bc ----------------
__global__ void gcn_final(const float* __restrict__ gsum, const float* __restrict__ cnt,
                          const float* __restrict__ Wc, const float* __restrict__ bc,
                          float* __restrict__ out) {
  int g = blockIdx.x, c = threadIdx.x;   // 64 x 20
  float inv = 1.0f / fmaxf(cnt[g], 1.0f);
  float acc = bc[c];
  for (int k = 0; k < 128; ++k)
    acc += gsum[(size_t)g * 128 + k] * inv * Wc[(size_t)k * 20 + c];
  out[(size_t)g * 20 + c] = acc;
}

extern "C" void kernel_launch(void* const* d_in, const int* in_sizes, int n_in,
                              void* d_out, int out_size, void* d_ws, size_t ws_size,
                              hipStream_t stream) {
  const float* x   = (const float*)d_in[0];
  const float* W1  = (const float*)d_in[1];
  const float* b1  = (const float*)d_in[2];
  const float* W2  = (const float*)d_in[3];
  const float* b2  = (const float*)d_in[4];
  const float* Wc  = (const float*)d_in[5];
  const float* bc  = (const float*)d_in[6];
  const int* eidx  = (const int*)d_in[7];
  const int* batch = (const int*)d_in[8];

  const int N = in_sizes[8];          // 50000 nodes
  const int E = in_sizes[7] / 2;      // 1.6M edges
  const int K1 = 768;
  const int* src = eidx;
  const int* dst = eidx + E;

  // ---- workspace carve-out ----
  char* base = (char*)d_ws;
  size_t off = 0;
  auto carve = [&](size_t bytes) {
    char* p = base + off;
    off = (off + bytes + 255) & ~(size_t)255;
    return p;
  };
  float*        deg    = (float*)carve((size_t)N * 4);
  float*        dinv   = (float*)carve((size_t)N * 4);
  unsigned int* B1frag = (unsigned int*)carve((size_t)(K1 / 32) * 8 * 32 * 8 * 4);
  unsigned int* B2frag = (unsigned int*)carve((size_t)(128 / 32) * 8 * 32 * 8 * 4);
  float*        hbuf   = (float*)carve((size_t)N * 128 * 4);   // h1 then h2
  float*        agg    = (float*)carve((size_t)N * 128 * 4);   // reused per layer
  __bf16*       hbf    = (__bf16*)carve((size_t)N * 128 * 2);  // relu(layer1) bf16
  float*        hrelu2 = (float*)carve((size_t)N * 128 * 4);   // relu(layer2) f32
  float*        gsum   = (float*)carve((size_t)64 * 128 * 4);
  float*        cnt    = (float*)carve((size_t)64 * 4);
  (void)ws_size; (void)n_in; (void)out_size;

  // ---- weight fragment packing (tiny, once per call) ----
  gcn_pack_B<<<(K1 / 32) * 8 * 32 / 256, 256, 0, stream>>>(W1, B1frag, K1);
  gcn_pack_B<<<(128 / 32) * 8 * 32 / 256, 256, 0, stream>>>(W2, B2frag, 128);

  // ---- degrees ----
  gcn_deg_init<<<(N + 255) / 256, 256, 0, stream>>>(deg, N);
  gcn_deg_count<<<(E + 255) / 256, 256, 0, stream>>>(dst, deg, E);
  gcn_dinv<<<(N + 255) / 256, 256, 0, stream>>>(deg, dinv, N);

  const int gemmBlocks = (N + 127) / 128;

  // ---- layer 1 ----
  gcn_gemm_wmma<768, true><<<gemmBlocks, 256, 0, stream>>>(x, B1frag, hbuf, N);
  hipMemsetAsync(agg, 0, (size_t)N * 128 * 4, stream);
  gcn_edge_scatter<<<(E + 7) / 8, 256, 0, stream>>>(hbuf, dinv, src, dst, agg, E);
  gcn_post<<<N, 128, 0, stream>>>(agg, hbuf, dinv, b1, hbf, nullptr);

  // ---- layer 2 ----
  gcn_gemm_wmma<128, false><<<gemmBlocks, 256, 0, stream>>>(hbf, B2frag, hbuf, N);
  hipMemsetAsync(agg, 0, (size_t)N * 128 * 4, stream);
  gcn_edge_scatter<<<(E + 7) / 8, 256, 0, stream>>>(hbuf, dinv, src, dst, agg, E);
  gcn_post<<<N, 128, 0, stream>>>(agg, hbuf, dinv, b2, nullptr, hrelu2);

  // ---- pool + classifier ----
  hipMemsetAsync(gsum, 0, (size_t)64 * 128 * 4, stream);
  hipMemsetAsync(cnt, 0, (size_t)64 * 4, stream);
  gcn_pool<<<N, 128, 0, stream>>>(hrelu2, batch, gsum, cnt);
  gcn_final<<<64, 20, 0, stream>>>(gsum, cnt, Wc, bc, (float*)d_out);
}
